// ROLLOUT_72954314489893
// MI455X (gfx1250) — matrix operand order repaired and checked by reading, compiled, not run on hardware
//
#include <hip/hip_runtime.h>
#include <hip/hip_bf16.h>

// Problem constants (from reference): V=32000, B=64, E=64, H=256, T=32
#define VV   32000
#define BB   64
#define EE   64
#define HH   256
#define TT   32
#define KX   (EE + HH)   // 320 = concat [x,h] K-dim for gate GEMM
#define G4   (4 * HH)    // 1024 = 4 gates

typedef __attribute__((ext_vector_type(16))) __bf16 v16bf;
typedef __attribute__((ext_vector_type(8)))  __bf16 v8bf;
typedef __attribute__((ext_vector_type(8)))  float  v8f;

union Frag16 { v16bf v; v8bf h[2]; };

__device__ __forceinline__ unsigned short f2bf(float f) {
    unsigned u = __float_as_uint(f);
    unsigned r = u + 0x7FFFu + ((u >> 16) & 1u);   // round-to-nearest-even
    return (unsigned short)(r >> 16);
}

// A-fragment (16x32 bf16): lane m = lane%16 is the row; per-lane K coverage is
// two contiguous 8-element chunks: [8*half, +8) and [16+8*half, +8).
__device__ __forceinline__ v16bf load_afrag(const unsigned short* rowk, int half) {
    Frag16 f;
    f.h[0] = *(const v8bf*)(rowk + 8 * half);
    f.h[1] = *(const v8bf*)(rowk + 16 + 8 * half);
    return f.v;
}

// B-fragment (32x16 bf16): lane n = lane%16 is the column; per-lane K coverage
// is two 8-element chunks [8*half, +8) and [16+8*half, +8) (matches A).
__device__ __forceinline__ v16bf load_bfrag(const unsigned short* colk, int half) {
    Frag16 f;
    f.h[0] = *(const v8bf*)(colk + 8 * half);
    f.h[1] = *(const v8bf*)(colk + 16 + 8 * half);
    return f.v;
}

__device__ __forceinline__ unsigned long long shflxor64(unsigned long long v, int m) {
    unsigned lo = (unsigned)v, hi = (unsigned)(v >> 32);
    lo = __shfl_xor(lo, m, 16);   // width 16: keep within each 16-lane half of wave32
    hi = __shfl_xor(hi, m, 16);
    return ((unsigned long long)hi << 32) | lo;
}

// ---------------- one-time prep kernels (bf16 weight packing) ----------------

// WcatT[n*320 + k], n in [0,1024): gate-major columns [i|f|o|c], K = [x(64) | h(256)]
__global__ __launch_bounds__(256) void pack_wcat_k(
    const float* __restrict__ Wi, const float* __restrict__ Ui,
    const float* __restrict__ Wf, const float* __restrict__ Uf,
    const float* __restrict__ Wog, const float* __restrict__ Uog,
    const float* __restrict__ Wc, const float* __restrict__ Uc,
    unsigned short* __restrict__ WcatT) {
    int idx = blockIdx.x * 256 + threadIdx.x;          // 1024*320 = 327680
    int n = idx / KX, k = idx % KX;
    int g = n >> 8, j = n & 255;
    const float* W = (g == 0) ? Wi : (g == 1) ? Wf : (g == 2) ? Wog : Wc;
    const float* U = (g == 0) ? Ui : (g == 1) ? Uf : (g == 2) ? Uog : Uc;
    float v = (k < EE) ? W[k * HH + j] : U[(k - EE) * HH + j];
    WcatT[idx] = f2bf(v);
}

// WoT[n*256 + k] = bf16(Wo[k*32000 + n]) via LDS tile transpose
__global__ __launch_bounds__(256) void transpose_wo_k(
    const float* __restrict__ Wo, unsigned short* __restrict__ WoT) {
    __shared__ float tile[32][33];
    int tx = threadIdx.x & 31, ty = threadIdx.x >> 5;   // 32x8
    int bx = blockIdx.x, by = blockIdx.y;               // 1000 x 8
#pragma unroll
    for (int i = 0; i < 4; ++i) {
        int k = by * 32 + ty + i * 8;
        int n = bx * 32 + tx;
        tile[ty + i * 8][tx] = Wo[(size_t)k * VV + n];
    }
    __syncthreads();
#pragma unroll
    for (int i = 0; i < 4; ++i) {
        int n = bx * 32 + ty + i * 8;
        int k = by * 32 + tx;
        WoT[(size_t)n * HH + k] = f2bf(tile[tx][ty + i * 8]);
    }
}

__global__ __launch_bounds__(256) void pack_emb_k(
    const float* __restrict__ emb, unsigned short* __restrict__ out) {
    int idx = blockIdx.x * 256 + threadIdx.x;           // V*E = 2,048,000
    out[idx] = f2bf(emb[idx]);
}

__global__ __launch_bounds__(256) void init_state_k(
    const int* __restrict__ start_token, const unsigned short* __restrict__ embB,
    unsigned short* __restrict__ xh, float* __restrict__ cst,
    unsigned long long* __restrict__ am) {
    int idx = blockIdx.x * 256 + threadIdx.x;           // 80 blocks -> 20480
    if (idx < BB * KX) {
        int b = idx / KX, k = idx % KX;
        xh[idx] = (k < EE) ? embB[start_token[b] * EE + k] : (unsigned short)0;
    }
    if (idx < BB * HH) cst[idx] = 0.f;
    if (idx < BB) am[idx] = 0ull;
}

// ---------------- per-step kernels ----------------

// z[64,1024] = xh[64,320] @ Wcat[320,1024]; 8 waves/block, wave = (ntile,mtile)
__global__ __launch_bounds__(256) void gates_gemm_k(
    const unsigned short* __restrict__ xh, const unsigned short* __restrict__ WcatT,
    float* __restrict__ gates) {
    int wave = threadIdx.x >> 5, lane = threadIdx.x & 31;
    int half = lane >> 4, ln = lane & 15;
    int mtile = wave & 3;
    int ntile = blockIdx.x * 2 + (wave >> 2);           // 32 blocks -> 64 ntiles
    const unsigned short* arow = xh + (mtile * 16 + ln) * KX;
    const unsigned short* brow = WcatT + (ntile * 16 + ln) * KX;
    v8f acc = {};
#pragma unroll
    for (int kt = 0; kt < KX / 32; ++kt) {
        v16bf a = load_afrag(arow + kt * 32, half);
        v16bf b = load_bfrag(brow + kt * 32, half);
        acc = __builtin_amdgcn_wmma_f32_16x16x32_bf16(false, a, false, b,
                                                      (short)0, acc, false, false);
    }
#pragma unroll
    for (int g = 0; g < 8; ++g)
        gates[(mtile * 16 + g + 8 * half) * G4 + ntile * 16 + ln] = acc[g];
}

__global__ __launch_bounds__(256) void lstm_pointwise_k(
    const float* __restrict__ gates, const float* __restrict__ bi,
    const float* __restrict__ bfg, const float* __restrict__ bog,
    const float* __restrict__ bc, float* __restrict__ cst,
    unsigned short* __restrict__ xh) {
    int idx = blockIdx.x * 256 + threadIdx.x;           // 64 blocks -> 16384
    int b = idx >> 8, j = idx & 255;
    const float* z = gates + b * G4;
    float ig = 1.f / (1.f + __expf(-(z[j] + bi[j])));
    float fg = 1.f / (1.f + __expf(-(z[256 + j] + bfg[j])));
    float og = 1.f / (1.f + __expf(-(z[512 + j] + bog[j])));
    float ch = tanhf(z[768 + j] + bc[j]);
    float cn = fg * cst[idx] + ig * ch;
    cst[idx] = cn;
    xh[b * KX + EE + j] = f2bf(og * tanhf(cn));         // h in bf16, ready as next A
}

// logits[64,32000] = h @ Wo + bo, fused row-argmax. Wave owns one 16-col strip,
// iterates 4 M-tiles x 8 K-steps (32 WMMA/wave), never spills logits to memory.
__global__ __launch_bounds__(256) void logits_argmax_k(
    const unsigned short* __restrict__ xh, const unsigned short* __restrict__ WoT,
    const float* __restrict__ bo, unsigned long long* __restrict__ am) {
    __shared__ unsigned long long sm[BB];
    if (threadIdx.x < BB) sm[threadIdx.x] = 0ull;
    __syncthreads();

    int wave = threadIdx.x >> 5, lane = threadIdx.x & 31;
    int half = lane >> 4, ln = lane & 15;
    int ntile = blockIdx.x * 8 + wave;                  // 250 blocks -> 2000 ntiles
    int ncol = ntile * 16 + ln;
    const unsigned short* brow = WoT + (size_t)ncol * HH;

    v8f acc[4] = {{}, {}, {}, {}};
#pragma unroll
    for (int kt = 0; kt < HH / 32; ++kt) {
        int kk = kt * 32;
        v16bf b = load_bfrag(brow + kk, half);
        if (kt < 7) __builtin_prefetch(brow + kk + 32, 0, 1);  // global_prefetch_b8
#pragma unroll
        for (int mt = 0; mt < 4; ++mt) {
            v16bf a = load_afrag(xh + (mt * 16 + ln) * KX + EE + kk, half);
            acc[mt] = __builtin_amdgcn_wmma_f32_16x16x32_bf16(false, a, false, b,
                                                              (short)0, acc[mt],
                                                              false, false);
        }
    }

    float bocol = bo[ncol];
#pragma unroll
    for (int mt = 0; mt < 4; ++mt) {
#pragma unroll
        for (int g = 0; g < 8; ++g) {
            float v = acc[mt][g] + bocol;
            unsigned k = __float_as_uint(v);
            k = (k & 0x80000000u) ? ~k : (k | 0x80000000u);  // order-preserving key
            unsigned long long p =
                ((unsigned long long)k << 32) | (unsigned)(~(unsigned)ncol);
#pragma unroll
            for (int m = 1; m < 16; m <<= 1) {               // max across 16 columns
                unsigned long long q = shflxor64(p, m);
                if (q > p) p = q;
            }
            if (ln == 0)
                atomicMax(&sm[mt * 16 + g + 8 * half], p);   // ds_max_u64
        }
    }
    __syncthreads();
    if (threadIdx.x < BB)
        atomicMax(&am[threadIdx.x], sm[threadIdx.x]);        // global_atomic_max_u64
}

__global__ __launch_bounds__(64) void select_tok_k(
    unsigned long long* __restrict__ am, const int* __restrict__ input_x,
    const int* __restrict__ given_num, const unsigned short* __restrict__ embB,
    unsigned short* __restrict__ xh, int* __restrict__ out, int t) {
    int b = blockIdx.x, e = threadIdx.x;
    unsigned long long p = am[b];
    int samp = (int)(~(unsigned)(p & 0xFFFFFFFFull));        // recover argmax column
    int tok = (t < *given_num) ? input_x[b * TT + t] : samp;
    __syncthreads();
    if (e == 0) { am[b] = 0ull; out[b * TT + t] = tok; }     // reset for next step
    xh[b * KX + e] = embB[tok * EE + e];                     // x_next = emb[tok]
}

// ---------------- host orchestration ----------------

extern "C" void kernel_launch(void* const* d_in, const int* in_sizes, int n_in,
                              void* d_out, int out_size, void* d_ws, size_t ws_size,
                              hipStream_t stream) {
    const int*   input_x     = (const int*)d_in[0];
    const int*   given_num   = (const int*)d_in[1];
    const int*   start_token = (const int*)d_in[2];
    const float* g_emb       = (const float*)d_in[3];
    const float* Wi  = (const float*)d_in[4];  const float* Ui  = (const float*)d_in[5];
    const float* bi  = (const float*)d_in[6];
    const float* Wf  = (const float*)d_in[7];  const float* Uf  = (const float*)d_in[8];
    const float* bfg = (const float*)d_in[9];
    const float* Wog = (const float*)d_in[10]; const float* Uog = (const float*)d_in[11];
    const float* bog = (const float*)d_in[12];
    const float* Wc  = (const float*)d_in[13]; const float* Uc  = (const float*)d_in[14];
    const float* bc  = (const float*)d_in[15];
    const float* Wo  = (const float*)d_in[16]; const float* bo  = (const float*)d_in[17];
    int* out = (int*)d_out;

    char* ws = (char*)d_ws;
    size_t off = 0;
    auto take = [&](size_t bytes) { char* p = ws + off; off = (off + bytes + 255) & ~(size_t)255; return p; };
    unsigned short*     WcatT = (unsigned short*)take((size_t)G4 * KX * 2);     // 640 KB
    unsigned short*     WoT   = (unsigned short*)take((size_t)VV * HH * 2);     // 16.4 MB
    unsigned short*     embB  = (unsigned short*)take((size_t)VV * EE * 2);     // 4.1 MB
    unsigned short*     xh    = (unsigned short*)take((size_t)BB * KX * 2);
    float*              gates = (float*)take((size_t)BB * G4 * 4);
    float*              cst   = (float*)take((size_t)BB * HH * 4);
    unsigned long long* am    = (unsigned long long*)take((size_t)BB * 8);
    (void)ws_size; (void)in_sizes; (void)n_in; (void)out_size;

    // one-time per call: bf16 weight packing
    pack_wcat_k<<<(G4 * KX) / 256, 256, 0, stream>>>(Wi, Ui, Wf, Uf, Wog, Uog, Wc, Uc, WcatT);
    transpose_wo_k<<<dim3(VV / 32, HH / 32), 256, 0, stream>>>(Wo, WoT);
    pack_emb_k<<<(VV * EE) / 256, 256, 0, stream>>>(g_emb, embB);
    init_state_k<<<(BB * KX + 255) / 256, 256, 0, stream>>>(start_token, embB, xh, cst, am);

    for (int t = 0; t < TT; ++t) {
        gates_gemm_k<<<32, 256, 0, stream>>>(xh, WcatT, gates);
        lstm_pointwise_k<<<BB, 256, 0, stream>>>(gates, bi, bfg, bog, bc, cst, xh);
        logits_argmax_k<<<(VV / 16) / 8, 256, 0, stream>>>(xh, WoT, bo, am);
        select_tok_k<<<BB, 64, 0, stream>>>(am, input_x, given_num, embB, xh, out, t);
    }
}
